// Decoder_42992622633145
// MI455X (gfx1250) — matrix operand order repaired and checked by reading, not compiled, not run on hardware
//
#include <hip/hip_runtime.h>

typedef __attribute__((ext_vector_type(16))) __bf16 v16bf;
typedef __attribute__((ext_vector_type(8)))  float  v8f;
typedef __attribute__((ext_vector_type(4)))  unsigned int v4u;
typedef __attribute__((ext_vector_type(8)))  int v8i;
typedef __attribute__((ext_vector_type(4)))  int v4i;
typedef unsigned short u16;

// Problem constants
constexpr int kDim   = 1024;
constexpr int kHeads = 16;
constexpr int kDh    = 64;
constexpr int kInner = 1024;
constexpr int kMlp   = 4096;
constexpr int kB     = 4;
constexpr int kT     = 1024;
constexpr int kS     = 512;       // PHASE+1
constexpr float kScale = 0.125f;  // 64^-0.5

__device__ __forceinline__ u16 f2bf(float f) {
  unsigned int u = __float_as_uint(f);
  unsigned int r = u + 0x7FFFu + ((u >> 16) & 1u);   // round-to-nearest-even
  return (u16)(r >> 16);
}

#if __has_builtin(__builtin_amdgcn_tensor_load_to_lds) && \
    __has_builtin(__builtin_amdgcn_s_wait_tensorcnt)
#define USE_TDM 1
#else
#define USE_TDM 0
#endif

#define BM 128
#define BN 128
#define BK 32
#define LDSW 40   // bf16 elems per LDS row (32 + 8 pad -> 80B, breaks bank alias)

#if USE_TDM
// Issue one TDM 2-D tile load: 128 rows x 32 bf16, row pitch stride_elems,
// LDS padding 4 DWORDs every 16 DWORDs (-> 40 bf16 LDS pitch).
// dim0_rem/dim1_rem: remaining tensor extent from tile origin (OOB reads -> 0).
__device__ __forceinline__ void tdm_load_tile(const u16* gptr, unsigned int lds_off,
                                              unsigned int dim0_rem, unsigned int dim1_rem,
                                              unsigned int stride_elems) {
  unsigned long long ga = (unsigned long long)(uintptr_t)gptr;
  v4u g0;
  g0[0] = 1u;                                            // count=1, user mode
  g0[1] = lds_off;                                       // lds_addr (bytes)
  g0[2] = (unsigned int)(ga & 0xFFFFFFFFull);            // global_addr[31:0]
  g0[3] = (unsigned int)((ga >> 32) & 0x01FFFFFFull)     // global_addr[56:32]
          | (2u << 30);                                  // type=2 ("image")
  v8i g1;
  g1[0] = (int)((1u << 16)      // data_size = 1 -> 2 bytes
              | (1u << 20)      // pad_enable
              | (3u << 22)      // pad_interval: 16 DWORDs
              | (3u << 25));    // pad_amount:   4 DWORDs (16B)
  g1[1] = (int)((dim0_rem & 0xFFFFu) << 16);                         // tensor_dim0 lo16
  g1[2] = (int)((dim0_rem >> 16) | ((dim1_rem & 0xFFFFu) << 16));    // dim0 hi16 | dim1 lo16
  g1[3] = (int)((dim1_rem >> 16) | (32u << 16));                     // dim1 hi16 | tile_dim0=32
  g1[4] = (int)(128u);                                               // tile_dim1=128, tile_dim2=0
  g1[5] = (int)stride_elems;                                         // tensor_dim0_stride lo32
  g1[6] = 0;                                                         // stride hi16 | dim1_stride lo
  g1[7] = 0;
  v4i z4 = {0, 0, 0, 0};
  v8i z8 = {0, 0, 0, 0, 0, 0, 0, 0};
  __builtin_amdgcn_tensor_load_to_lds(g0, g1, z4, z4, z8, 0);
}
#endif

// ---------------------------------------------------------------------------
// Generic batched GEMM: C[M][N] (+bias,+gelu) = A_bf16[M][K] x Bt_bf16[N][K]^T
// Bt is N-major (row n = K-vector for output column n).
// grid.z batches decompose as z = zb*zdiv + zh with independent strides.
// Block: 256 threads = 8 waves; tile 128x128x32; wave tile 32x64 (2x4 frags).
// Staging: double-buffered TDM (tensor_load_to_lds) when available.
// ---------------------------------------------------------------------------
__global__ __launch_bounds__(256) void k_gemm(
    const u16* __restrict__ A, long long sAb, long long sAh, int lda,
    const u16* __restrict__ Bt, long long sBb, long long sBh, int ldb,
    float* __restrict__ C, u16* __restrict__ Cb, long long sCb, long long sCh, int ldc,
    const float* __restrict__ bias, int M, int N, int K, int zdiv, int gelu)
{
  __shared__ u16 As[2][BM * LDSW];
  __shared__ u16 Bs[2][BN * LDSW];

  const int tid  = threadIdx.x;
  const int lane = tid & 31;
  const int wave = tid >> 5;
  const int wm   = wave >> 1;          // 4 wave rows
  const int wn   = wave & 1;           // 2 wave cols
  const int l16  = lane & 15;
  const int lhi  = (lane >> 4) & 1;

  const int z  = blockIdx.z;
  const int zb = z / zdiv, zh = z % zdiv;
  const u16* Ab = A  + (size_t)zb * sAb + (size_t)zh * sAh;
  const u16* Bb = Bt + (size_t)zb * sBb + (size_t)zh * sBh;
  const size_t coff = (size_t)zb * sCb + (size_t)zh * sCh;

  const int m0 = blockIdx.y * BM;
  const int n0 = blockIdx.x * BN;

  const v8f vzero = {0.f,0.f,0.f,0.f,0.f,0.f,0.f,0.f};
  v8f acc[2][4];
#pragma unroll
  for (int i = 0; i < 2; ++i)
#pragma unroll
    for (int j = 0; j < 4; ++j) acc[i][j] = vzero;

#if USE_TDM
  const unsigned int dim1A = (unsigned int)(M - m0);   // rows beyond -> zero-fill
  const unsigned int dim1B = (unsigned int)(N - n0);
  const unsigned int ldsA0 = (unsigned int)(uintptr_t)&As[0][0];
  const unsigned int ldsA1 = (unsigned int)(uintptr_t)&As[1][0];
  const unsigned int ldsB0 = (unsigned int)(uintptr_t)&Bs[0][0];
  const unsigned int ldsB1 = (unsigned int)(uintptr_t)&Bs[1][0];
  if (tid < 32) {   // one wave drives the Tensor Data Mover
    tdm_load_tile(Ab + (size_t)m0 * lda, ldsA0, (unsigned int)K, dim1A, (unsigned int)lda);
    tdm_load_tile(Bb + (size_t)n0 * ldb, ldsB0, (unsigned int)K, dim1B, (unsigned int)ldb);
  }
#endif

  for (int k0 = 0; k0 < K; k0 += BK) {
#if USE_TDM
    const int cur = (k0 / BK) & 1;
    if (tid < 32) __builtin_amdgcn_s_wait_tensorcnt(0);
    __syncthreads();                       // tiles[cur] visible to all waves
    if (tid < 32 && (k0 + BK) < K) {       // prefetch next tiles, overlapped with math
      tdm_load_tile(Ab + (size_t)m0 * lda + (k0 + BK), cur ? ldsA0 : ldsA1,
                    (unsigned int)(K - k0 - BK), dim1A, (unsigned int)lda);
      tdm_load_tile(Bb + (size_t)n0 * ldb + (k0 + BK), cur ? ldsB0 : ldsB1,
                    (unsigned int)(K - k0 - BK), dim1B, (unsigned int)ldb);
    }
    const u16* Asb = &As[cur][0];
    const u16* Bsb = &Bs[cur][0];
#else
    // ---- manual staging fallback: 16B chunks, 2 per thread per tile
    for (int i = tid; i < BM * 4; i += 256) {
      int r = i >> 2, c = (i & 3) << 3;
      uint4 val{0u, 0u, 0u, 0u};
      int gr = m0 + r;
      if (gr < M) val = *(const uint4*)(Ab + (size_t)gr * lda + (k0 + c));
      *(uint4*)(&As[0][r * LDSW + c]) = val;
    }
    for (int i = tid; i < BN * 4; i += 256) {
      int r = i >> 2, c = (i & 3) << 3;
      uint4 val{0u, 0u, 0u, 0u};
      int gn = n0 + r;
      if (gn < N) val = *(const uint4*)(Bb + (size_t)gn * ldb + (k0 + c));
      *(uint4*)(&Bs[0][r * LDSW + c]) = val;
    }
    __syncthreads();
    const u16* Asb = &As[0][0];
    const u16* Bsb = &Bs[0][0];
#endif

    // ---- fragments (ISA 7.12.2 layouts)
    union U { uint4 u[2]; v16bf v; };
    U ua0, ua1, ub[4];
    {
      // A 16x32: lane<16 -> K{0..7,16..23}; lane>=16 -> K{8..15,24..31}
      int r0 = wm * 32 + l16;
      const uint4* p0 = (const uint4*)(&Asb[r0 * LDSW]);
      ua0.u[0] = p0[lhi];
      ua0.u[1] = p0[2 + lhi];
      const uint4* p1 = (const uint4*)(&Asb[(r0 + 16) * LDSW]);
      ua1.u[0] = p1[lhi];
      ua1.u[1] = p1[2 + lhi];
    }
#pragma unroll
    for (int ni = 0; ni < 4; ++ni) {
      // B 32x16: lanes 0-15 hold K{0..7,16..23}, lanes 16-31 hold K{8..15,24..31} (col = l16)
      int nrow = wn * 64 + ni * 16 + l16;
      const uint4* q = (const uint4*)(&Bsb[nrow * LDSW]);
      ub[ni].u[0] = q[lhi];
      ub[ni].u[1] = q[2 + lhi];
    }
#pragma unroll
    for (int ni = 0; ni < 4; ++ni) {
      acc[0][ni] = __builtin_amdgcn_wmma_f32_16x16x32_bf16(
          false, ua0.v, false, ub[ni].v, (short)0, acc[0][ni], false, false);
      acc[1][ni] = __builtin_amdgcn_wmma_f32_16x16x32_bf16(
          false, ua1.v, false, ub[ni].v, (short)0, acc[1][ni], false, false);
    }

#if USE_TDM
    // no trailing barrier needed: buffer[cur] is only overwritten by the issue
    // that follows the NEXT iteration's barrier, which all waves reach only
    // after finishing these reads.
#else
    __syncthreads();
    if (k0 + BK < K) {
      int pr = m0 + (tid >> 1);
      if (pr < M) __builtin_prefetch(Ab + (size_t)pr * lda + (k0 + BK), 0, 0);
      int pn = n0 + (tid >> 1);
      if (pn < N) __builtin_prefetch(Bb + (size_t)pn * ldb + (k0 + BK), 0, 0);
    }
#endif
  }

  // ---- epilogue: C/D layout VGPR e -> row e (lanes 0-15) / e+8 (lanes 16-31)
#pragma unroll
  for (int mi = 0; mi < 2; ++mi) {
#pragma unroll
    for (int ni = 0; ni < 4; ++ni) {
      int col = n0 + wn * 64 + ni * 16 + l16;
      if (col >= N) continue;
      float bv = bias ? bias[col] : 0.0f;
#pragma unroll
      for (int e = 0; e < 8; ++e) {
        int row = m0 + wm * 32 + mi * 16 + lhi * 8 + e;
        if (row >= M) continue;
        float v = acc[mi][ni][e] + bv;
        if (gelu) v = 0.5f * v * (1.0f + erff(v * 0.70710678118654752f));
        size_t off = coff + (size_t)row * ldc + col;
        if (C)  C[off]  = v;
        if (Cb) Cb[off] = f2bf(v);
      }
    }
  }
}

// ---------------------------------------------------------------------------
// Weight transpose + convert: in f32 [K][N] -> out bf16 [N][K]
// ---------------------------------------------------------------------------
__global__ void k_wt(const float* __restrict__ in, u16* __restrict__ out, int K, int N) {
  long long idx = (long long)blockIdx.x * 256 + threadIdx.x;
  long long total = (long long)K * N;
  if (idx >= total) return;
  int k = (int)(idx % K);
  long long n = idx / K;
  out[idx] = f2bf(in[(size_t)k * N + n]);
}

// f32 -> bf16 elementwise
__global__ void k_cvt(const float* __restrict__ in, u16* __restrict__ out, long long total) {
  long long idx = (long long)blockIdx.x * 256 + threadIdx.x;
  if (idx < total) out[idx] = f2bf(in[idx]);
}

// V head-transpose: in bf16 [B, S, H*64] -> out bf16 [B, H, 64, S]
__global__ void k_vt(const u16* __restrict__ in, u16* __restrict__ out, int S) {
  long long idx = (long long)blockIdx.x * 256 + threadIdx.x;
  long long total = (long long)kB * kHeads * kDh * S;
  if (idx >= total) return;
  int s = (int)(idx % S);
  long long r = idx / S;
  int d = (int)(r % kDh); r /= kDh;
  int h = (int)(r % kHeads);
  int b = (int)(r / kHeads);
  out[idx] = in[((size_t)b * S + s) * kInner + h * kDh + d];
}

// x0 = dec[...,None]*pe_w + pe_b + pos_emb ; writes f32 + bf16
__global__ void k_embed(const float* __restrict__ dec, const float* __restrict__ pw,
                        const float* __restrict__ pb, const float* __restrict__ pos,
                        float* __restrict__ x0, u16* __restrict__ x0b) {
  long long idx = (long long)blockIdx.x * 256 + threadIdx.x;
  long long total = (long long)kB * kT * kDim;
  if (idx >= total) return;
  int d = (int)(idx & (kDim - 1));
  long long bt = idx >> 10;
  float v = dec[bt] * pw[d] + pb[d] + pos[d];
  x0[idx] = v;
  x0b[idx] = f2bf(v);
}

// Row softmax with scale + optional key-pad mask; bf16 output.
// mask semantics: col j masked iff j>0 && encm[b*(L-1) + (j-1)] == 0.0f -> -1e9
__global__ __launch_bounds__(256) void k_softmax(const float* __restrict__ dots,
                                                 u16* __restrict__ probs,
                                                 const float* __restrict__ encm,
                                                 int L, int rows_per_b, float scale) {
  int row = blockIdx.x;
  int b = row / rows_per_b;
  const float* drow = dots + (size_t)row * L;
  u16* prow = probs + (size_t)row * L;
  int tid = threadIdx.x;
  int nit = L >> 8;          // 4 (L=1024) or 2 (L=512)
  float vals[4];
  float m = -3.4e38f;
  for (int i = 0; i < nit; ++i) {
    int j = tid + (i << 8);
    float v = drow[j] * scale;
    if (encm && j > 0 && encm[(size_t)b * (L - 1) + (j - 1)] == 0.0f) v = -1e9f;
    vals[i] = v;
    m = fmaxf(m, v);
  }
  __shared__ float red[256];
  red[tid] = m; __syncthreads();
  for (int st = 128; st > 0; st >>= 1) {
    if (tid < st) red[tid] = fmaxf(red[tid], red[tid + st]);
    __syncthreads();
  }
  m = red[0]; __syncthreads();
  float sum = 0.f;
  for (int i = 0; i < nit; ++i) { vals[i] = __expf(vals[i] - m); sum += vals[i]; }
  red[tid] = sum; __syncthreads();
  for (int st = 128; st > 0; st >>= 1) {
    if (tid < st) red[tid] += red[tid + st];
    __syncthreads();
  }
  float inv = 1.0f / red[0];
  for (int i = 0; i < nit; ++i) prow[tid + (i << 8)] = f2bf(vals[i] * inv);
}

// out = LayerNorm(a + r) * g + b ; writes f32 master + bf16 copy
__global__ __launch_bounds__(256) void k_add_ln(const float* __restrict__ a,
                                                const float* __restrict__ r,
                                                const float* __restrict__ g,
                                                const float* __restrict__ bt,
                                                float* __restrict__ out,
                                                u16* __restrict__ outb) {
  int row = blockIdx.x;
  int tid = threadIdx.x;
  const size_t base = (size_t)row * kDim;
  float h[4]; float s = 0.f, sq = 0.f;
#pragma unroll
  for (int i = 0; i < 4; ++i) {
    int j = tid + (i << 8);
    float v = a[base + j] + r[base + j];
    h[i] = v; s += v; sq += v * v;
  }
  __shared__ float red[256], red2[256];
  red[tid] = s; red2[tid] = sq; __syncthreads();
  for (int st = 128; st > 0; st >>= 1) {
    if (tid < st) { red[tid] += red[tid + st]; red2[tid] += red2[tid + st]; }
    __syncthreads();
  }
  float mean = red[0] * (1.0f / kDim);
  float var  = red2[0] * (1.0f / kDim) - mean * mean;
  float rstd = rsqrtf(var + 1e-5f);
#pragma unroll
  for (int i = 0; i < 4; ++i) {
    int j = tid + (i << 8);
    float v = (h[i] - mean) * rstd * g[j] + bt[j];
    out[base + j] = v;
    outb[base + j] = f2bf(v);
  }
}

// ---------------------------------------------------------------------------
extern "C" void kernel_launch(void* const* d_in, const int* in_sizes, int n_in,
                              void* d_out, int out_size, void* d_ws, size_t ws_size,
                              hipStream_t stream) {
  (void)in_sizes; (void)n_in; (void)out_size; (void)ws_size;
  const float* dec  = (const float*)d_in[0];
  const float* encO = (const float*)d_in[1];
  const float* encI = (const float*)d_in[2];
  const float* pe_w = (const float*)d_in[3];
  const float* pe_b = (const float*)d_in[4];
  const float* pos  = (const float*)d_in[5];

  char* p = (char*)d_ws;
  auto alloc = [&](size_t bytes) -> void* {
    void* r = (void*)p;
    p += (bytes + 255) & ~(size_t)255;
    return r;
  };

  const size_t NACT = (size_t)kB * kT * kDim;                 // 4M activations
  float* x0   = (float*)alloc(NACT * 4);
  float* x    = (float*)alloc(NACT * 4);
  float* tmpf = (float*)alloc(NACT * 4);                      // attn-out / ff2-out
  float* dots = (float*)alloc((size_t)kB * kHeads * kT * kT * 4);
  u16* x0b  = (u16*)alloc(NACT * 2);
  u16* xb   = (u16*)alloc(NACT * 2);
  u16* qb   = (u16*)alloc(NACT * 2);
  u16* kb   = (u16*)alloc(NACT * 2);
  u16* vb   = (u16*)alloc(NACT * 2);
  u16* vtb  = (u16*)alloc(NACT * 2);
  u16* ob   = (u16*)alloc(NACT * 2);
  u16* probs = (u16*)alloc((size_t)kB * kHeads * kT * kT * 2);
  u16* encb  = (u16*)alloc((size_t)kB * kS * kDim * 2);
  u16* h1b   = (u16*)alloc((size_t)kB * kT * kMlp * 2);
  u16* W[4][10];
  for (int L = 0; L < 4; ++L) {
    for (int i = 0; i < 8; ++i) W[L][i] = (u16*)alloc((size_t)kDim * kInner * 2);
    W[L][8] = (u16*)alloc((size_t)kDim * kMlp * 2);
    W[L][9] = (u16*)alloc((size_t)kMlp * kDim * 2);
  }

  auto gemm = [&](const u16* A, long long sAb, long long sAh, int lda,
                  const u16* Bt, long long sBb, long long sBh, int ldb,
                  float* C, u16* Cb, long long sCb, long long sCh, int ldc,
                  const float* bias, int M, int N, int K, int Z, int zdiv, int gelu) {
    dim3 g((N + BN - 1) / BN, (M + BM - 1) / BM, Z);
    k_gemm<<<g, 256, 0, stream>>>(A, sAb, sAh, lda, Bt, sBb, sBh, ldb,
                                  C, Cb, sCb, sCh, ldc, bias, M, N, K, zdiv, gelu);
  };

  // ---- weight transpose+convert (once per launch)
  for (int L = 0; L < 4; ++L) {
    int base = 6 + 20 * L;
    const int KN[10][2] = {
      {kDim, kInner}, {kDim, kInner}, {kDim, kInner}, {kInner, kDim},   // a1 q,k,v,o
      {kDim, kInner}, {kDim, kInner}, {kDim, kInner}, {kInner, kDim},   // a2 q,k,v,o
      {kDim, kMlp}, {kMlp, kDim}                                        // ff w1,w2
    };
    const int srcIdx[10] = {0, 1, 2, 3, 7, 8, 9, 10, 14, 16};
    for (int i = 0; i < 10; ++i) {
      long long tot = (long long)KN[i][0] * KN[i][1];
      k_wt<<<(unsigned)((tot + 255) / 256), 256, 0, stream>>>(
          (const float*)d_in[base + srcIdx[i]], W[L][i], KN[i][0], KN[i][1]);
    }
  }

  // ---- embedding + encoder bf16
  k_embed<<<(unsigned)((NACT + 255) / 256), 256, 0, stream>>>(dec, pe_w, pe_b, pos, x0, x0b);
  {
    long long tot = (long long)kB * kS * kDim;
    k_cvt<<<(unsigned)((tot + 255) / 256), 256, 0, stream>>>(encO, encb, tot);
  }

  const int M4 = kB * kT;       // 4096
  const int M2 = kB * kS;       // 2048
  const int Z  = kB * kHeads;   // 64

  for (int L = 0; L < 4; ++L) {
    int base = 6 + 20 * L;
    const float* bo1 = (const float*)d_in[base + 4];
    const float* g1  = (const float*)d_in[base + 5];
    const float* b1  = (const float*)d_in[base + 6];
    const float* bo2 = (const float*)d_in[base + 11];
    const float* g2  = (const float*)d_in[base + 12];
    const float* b2  = (const float*)d_in[base + 13];
    const float* fb1 = (const float*)d_in[base + 15];
    const float* fb2 = (const float*)d_in[base + 17];
    const float* gf  = (const float*)d_in[base + 18];
    const float* bf  = (const float*)d_in[base + 19];

    // ===== self-attention (Q=K=V=x0) =====
    gemm(x0b,0,0,kDim, W[L][0],0,0,kDim, nullptr,qb,0,0,kInner, nullptr, M4,kInner,kDim, 1,1,0);
    gemm(x0b,0,0,kDim, W[L][1],0,0,kDim, nullptr,kb,0,0,kInner, nullptr, M4,kInner,kDim, 1,1,0);
    gemm(x0b,0,0,kDim, W[L][2],0,0,kDim, nullptr,vb,0,0,kInner, nullptr, M4,kInner,kDim, 1,1,0);
    {
      long long tot = (long long)kB * kHeads * kDh * kT;
      k_vt<<<(unsigned)((tot + 255) / 256), 256, 0, stream>>>(vb, vtb, kT);
    }
    // dots[b,h,t,t'] = Q K^T  (K operand already N-major: [seq][d])
    gemm(qb, (long long)kT*kInner, kDh, kInner,
         kb, (long long)kT*kInner, kDh, kInner,
         dots, nullptr, (long long)kHeads*kT*kT, (long long)kT*kT, kT,
         nullptr, kT, kT, kDh, Z, kHeads, 0);
    k_softmax<<<(unsigned)(Z * kT), 256, 0, stream>>>(dots, probs, nullptr, kT, kHeads * kT, kScale);
    // o[b,t,h*d] = P V   (Vt is N-major: [d][seq])
    gemm(probs, (long long)kHeads*kT*kT, (long long)kT*kT, kT,
         vtb, (long long)kHeads*kDh*kT, (long long)kDh*kT, kT,
         nullptr, ob, (long long)kT*kInner, kDh, kInner,
         nullptr, kT, kDh, kT, Z, kHeads, 0);
    gemm(ob,0,0,kInner, W[L][3],0,0,kInner, tmpf,nullptr,0,0,kDim, bo1, M4,kDim,kInner, 1,1,0);
    k_add_ln<<<(unsigned)M4, 256, 0, stream>>>(tmpf, x0, g1, b1, x, xb);

    // ===== cross-attention (Q=x, K=V=enc_output, key-pad mask) =====
    gemm(xb,0,0,kDim, W[L][4],0,0,kDim, nullptr,qb,0,0,kInner, nullptr, M4,kInner,kDim, 1,1,0);
    gemm(encb,0,0,kDim, W[L][5],0,0,kDim, nullptr,kb,0,0,kInner, nullptr, M2,kInner,kDim, 1,1,0);
    gemm(encb,0,0,kDim, W[L][6],0,0,kDim, nullptr,vb,0,0,kInner, nullptr, M2,kInner,kDim, 1,1,0);
    {
      long long tot = (long long)kB * kHeads * kDh * kS;
      k_vt<<<(unsigned)((tot + 255) / 256), 256, 0, stream>>>(vb, vtb, kS);
    }
    gemm(qb, (long long)kT*kInner, kDh, kInner,
         kb, (long long)kS*kInner, kDh, kInner,
         dots, nullptr, (long long)kHeads*kT*kS, (long long)kT*kS, kS,
         nullptr, kT, kS, kDh, Z, kHeads, 0);
    k_softmax<<<(unsigned)(Z * kT), 256, 0, stream>>>(dots, probs, encI, kS, kHeads * kT, kScale);
    gemm(probs, (long long)kHeads*kT*kS, (long long)kT*kS, kS,
         vtb, (long long)kHeads*kDh*kS, (long long)kDh*kS, kS,
         nullptr, ob, (long long)kT*kInner, kDh, kInner,
         nullptr, kT, kDh, kS, Z, kHeads, 0);
    gemm(ob,0,0,kInner, W[L][7],0,0,kInner, tmpf,nullptr,0,0,kDim, bo2, M4,kDim,kInner, 1,1,0);
    k_add_ln<<<(unsigned)M4, 256, 0, stream>>>(tmpf, x, g2, b2, x, xb);

    // ===== feed-forward =====
    gemm(xb,0,0,kDim, W[L][8],0,0,kDim, nullptr,h1b,0,0,kMlp, fb1, M4,kMlp,kDim, 1,1, 1 /*gelu*/);
    gemm(h1b,0,0,kMlp, W[L][9],0,0,kMlp, tmpf,nullptr,0,0,kDim, fb2, M4,kDim,kMlp, 1,1,0);
    k_add_ln<<<(unsigned)M4, 256, 0, stream>>>(tmpf, x, gf, bf, x, xb);
  }

  (void)hipMemcpyAsync(d_out, x, NACT * sizeof(float), hipMemcpyDeviceToDevice, stream);
}
